// MaskableLSTM_65695819759830
// MI455X (gfx1250) — matrix operand order repaired and checked by reading, not compiled, not run on hardware
//
#include <hip/hip_runtime.h>

namespace {
constexpr int kB = 64;          // batch
constexpr int kT = 256;         // time
constexpr int kD = 512;         // input feature
constexpr int kH = 512;         // hidden
constexpr int kG = 4 * kH;      // 2048 gate width (i,f,g,o)
constexpr int kM = kB * kT;     // 16384 rows, row index = b*T + t
}

typedef __bf16 bf16t;
typedef bf16t v16bf __attribute__((ext_vector_type(16)));
typedef bf16t v8bf  __attribute__((ext_vector_type(8)));
typedef float v8f   __attribute__((ext_vector_type(8)));

// ---------------------------------------------------------------------------
// WMMA fragment loaders (wave32, 16x16x32 bf16), contiguous K order, same for A and B.
// A (16x32, MxK): lane L holds row M=L&15; elems 0..15 = K[kb..kb+16),
//                 kb = k0 + (L>>4)*16.
// B (32x16, KxN): lane L holds col N=L&15; elems 0..15 = K[k0+(L>>4)*16 ..+16)
//                 (i.e. 16 contiguous K of row N of W, W stored [N,K] row-major).
// ---------------------------------------------------------------------------
__device__ __forceinline__ v16bf load_a_frag(const bf16t* base, int ld,
                                             int m0, int k0, int lane) {
  const int m  = m0 + (lane & 15);
  const int kb = k0 + ((lane >> 4) << 4);
  const bf16t* p = base + (size_t)m * ld + kb;
  return *(const v16bf*)(p);
}

__device__ __forceinline__ v16bf load_b_frag(const bf16t* base, int ld,
                                             int n0, int k0, int lane) {
  const int n  = n0 + (lane & 15);
  const int kb = k0 + ((lane >> 4) << 4);
  return *(const v16bf*)(base + (size_t)n * ld + kb);
}

__device__ __forceinline__ float sigf(float x) {
  return 1.0f / (1.0f + __expf(-x));
}

// ---------------------------------------------------------------------------
// dtype conversion f32 -> bf16 (also used to (re)seed the h ping buffer)
// ---------------------------------------------------------------------------
__global__ void cvt_f32_bf16_kernel(const float* __restrict__ src,
                                    bf16t* __restrict__ dst, int n) {
  int i = blockIdx.x * blockDim.x + threadIdx.x;
  const int stride = gridDim.x * blockDim.x;
  for (; i < n; i += stride) dst[i] = (bf16t)src[i];
}

__global__ void zero_u32_kernel(unsigned* __restrict__ p, int n) {
  const int i = blockIdx.x * blockDim.x + threadIdx.x;
  if (i < n) p[i] = 0u;
}

// ---------------------------------------------------------------------------
// Batched input projection: G[m, n] = bf16( X[m,:] . W[n,:] + bias[n] )
// X: [kM, 512] bf16, W: [2048, 512] bf16, G: [kM, 2048] bf16.
// Wave tiling: 16 rows x 64 cols (4 C tiles share one A fragment per k-step).
// ---------------------------------------------------------------------------
__global__ void __launch_bounds__(256)
gemm_xw_bias_kernel(const bf16t* __restrict__ X, const bf16t* __restrict__ W,
                    const float* __restrict__ bias, bf16t* __restrict__ G) {
  const int lane = threadIdx.x & 31;
  const int wave = blockIdx.x * (blockDim.x >> 5) + (threadIdx.x >> 5);
  const int cg = wave & 31;   // 32 column groups of 64
  const int rt = wave >> 5;   // 1024 row tiles of 16
  const int m0 = rt << 4;
  const int n0 = cg << 6;

  v8f acc[4] = {};
  for (int kk = 0; kk < 512; kk += 32) {
    v16bf a = load_a_frag(X, 512, m0, kk, lane);
#pragma unroll
    for (int gi = 0; gi < 4; ++gi) {
      v16bf b = load_b_frag(W, 512, n0 + (gi << 4), kk, lane);
      acc[gi] = __builtin_amdgcn_wmma_f32_16x16x32_bf16(
          false, a, false, b, (short)0, acc[gi], false, false);
    }
  }

  const int cl = lane & 15;
  const int rh = (lane >> 4) << 3;
#pragma unroll
  for (int gi = 0; gi < 4; ++gi) {
    const int n = n0 + (gi << 4) + cl;
    const float bv = bias[n];
#pragma unroll
    for (int r = 0; r < 8; ++r) {
      const int m = m0 + rh + r;
      G[(size_t)m * kG + n] = (bf16t)(acc[gi][r] + bv);
    }
  }
}

// ---------------------------------------------------------------------------
// Persistent recurrent scan. Grid = 64 WGs x 128 threads:
//   dir = blockIdx.x>>5 (0=fwd,1=bwd), wg = blockIdx.x&31 owns hidden cols
//   [wg*16, wg*16+16). Wave w (0..3) owns batch rows [16w,16w+16) and the
//   4 gate tiles (i,f,g,o) for that (rows x 16 hidden cols) patch, so the
//   cell state c lives in the accumulator register layout for all 256 steps.
//   Whh slice (4 gates x 16 rows x 512 K = 64KB bf16) is staged in LDS once.
//   h state ping-pongs between two 64KB global buffers; a per-direction
//   device-scope barrier separates timesteps.
// ---------------------------------------------------------------------------
struct DirArgs {
  const bf16t* gin;   // [kM, kG]  precomputed x@Wih^T + b (bf16)
  const bf16t* whh;   // [kG, kH]  bf16 recurrent weights
  bf16t* hping;       // [2][kB*kH] ping-pong hidden state (bf16)
  bf16t* ybf;         // [kM, kH] bf16 layer output (next layer input) or null
  float* yf32;        // d_out + dir*kH (row stride 2H) or null
  float* hN;          // final hidden dest (row stride 2H)
  float* cN;          // final cell dest (row stride 2H)
  unsigned* bar;      // barrier counter for this direction
  int reverse;        // 0 = forward scan, 1 = reverse scan
};
struct RecArgs {
  DirArgs dir[2];
  const float* masks; // [kB, kT]
  const float* h0;    // [kB, kH]
  const float* c0;    // [kB, kH]
};

__global__ void __launch_bounds__(128)
lstm_scan_kernel(RecArgs A) {
  constexpr int NWG = 32;
  extern __shared__ bf16t w_lds[];          // [64 rows][512 K] = 64KB
  const int d    = blockIdx.x >> 5;
  const int wg   = blockIdx.x & 31;
  const DirArgs da = A.dir[d];
  const int lane = threadIdx.x & 31;
  const int wave = threadIdx.x >> 5;
  const int m0   = wave << 4;               // batch-row tile
  const int hc0  = wg << 4;                 // hidden column base

  // Stage Whh rows {gi*512 + hc0 + [0,16)} x all K into LDS (once).
  for (int idx = threadIdx.x; idx < (64 * 512) / 8; idx += blockDim.x) {
    const int row = idx >> 6;               // 64 uint4 per 512-elem row
    const int c8  = idx & 63;
    const int gi  = row >> 4;
    const int lc  = row & 15;
    uint4 v = *(const uint4*)(da.whh +
        (size_t)(gi * 512 + hc0 + lc) * 512 + (c8 << 3));
    *(uint4*)(w_lds + (size_t)row * 512 + (c8 << 3)) = v;
  }
  __syncthreads();

  const int col = hc0 + (lane & 15);        // hidden column in [0,512)
  const int rh  = (lane >> 4) << 3;
  float h0f[8], c0f[8], cst[8];
#pragma unroll
  for (int r = 0; r < 8; ++r) {
    const int b = m0 + rh + r;
    h0f[r] = A.h0[b * kH + col];
    c0f[r] = A.c0[b * kH + col];
    cst[r] = c0f[r];
  }

  unsigned gen = 0;
  for (int step = 0; step < kT; ++step) {
    const int t = da.reverse ? (kT - 1 - step) : step;
    const bf16t* hread = da.hping + (size_t)(step & 1) * (kB * kH);
    bf16t* hwrite = da.hping + (size_t)((step + 1) & 1) * (kB * kH);

    // Prime accumulators with precomputed input projection + bias.
    v8f acc[4];
#pragma unroll
    for (int gi = 0; gi < 4; ++gi) {
#pragma unroll
      for (int r = 0; r < 8; ++r) {
        const int b = m0 + rh + r;
        acc[gi][r] = (float)da.gin[(size_t)(b * kT + t) * kG + gi * kH + col];
      }
    }
    // gates += h_prev @ Whh^T  (A from global ping buffer, B from LDS)
    for (int kk = 0; kk < kH; kk += 32) {
      v16bf a = load_a_frag(hread, kH, m0, kk, lane);
#pragma unroll
      for (int gi = 0; gi < 4; ++gi) {
        v16bf b = load_b_frag(w_lds, 512, gi << 4, kk, lane);
        acc[gi] = __builtin_amdgcn_wmma_f32_16x16x32_bf16(
            false, a, false, b, (short)0, acc[gi], false, false);
      }
    }
    // Cell update + valid-length masking, entirely in registers.
#pragma unroll
    for (int r = 0; r < 8; ++r) {
      const int b = m0 + rh + r;
      const float m  = A.masks[b * kT + t];
      const float ig = sigf(acc[0][r]);
      const float fg = sigf(acc[1][r]);
      const float gg = tanhf(acc[2][r]);
      const float og = sigf(acc[3][r]);
      float c2 = fg * cst[r] + ig * gg;
      float h2 = og * tanhf(c2);
      h2 = h2 * m + h0f[r] * (1.0f - m);
      c2 = c2 * m + c0f[r] * (1.0f - m);
      cst[r] = c2;
      hwrite[b * kH + col] = (bf16t)h2;
      if (da.ybf)  da.ybf[(size_t)(b * kT + t) * kH + col] = (bf16t)h2;
      if (da.yf32) da.yf32[(size_t)(b * kT + t) * (2 * kH) + col] = h2;
      if (step == kT - 1) {
        da.hN[b * (2 * kH) + col] = h2;
        da.cN[b * (2 * kH) + col] = c2;
      }
    }
    // Device-scope barrier among this direction's 32 workgroups.
    __threadfence();
    __syncthreads();
    if (threadIdx.x == 0) {
      const unsigned target = (gen + 1) * NWG;
      __hip_atomic_fetch_add(da.bar, 1u, __ATOMIC_ACQ_REL,
                             __HIP_MEMORY_SCOPE_AGENT);
      while (__hip_atomic_load(da.bar, __ATOMIC_ACQUIRE,
                               __HIP_MEMORY_SCOPE_AGENT) < target) {
        __builtin_amdgcn_s_sleep(2);
      }
    }
    __syncthreads();
    __threadfence();
    ++gen;
  }
}

// ---------------------------------------------------------------------------
// Host orchestration.
// ---------------------------------------------------------------------------
extern "C" void kernel_launch(void* const* d_in, const int* in_sizes, int n_in,
                              void* d_out, int out_size, void* d_ws,
                              size_t ws_size, hipStream_t stream) {
  (void)in_sizes; (void)n_in; (void)out_size; (void)ws_size;
  const float* x     = (const float*)d_in[0];
  const float* masks = (const float*)d_in[1];
  const float* h0    = (const float*)d_in[2];
  const float* c0    = (const float*)d_in[3];
  float* out = (float*)d_out;

  // Workspace carve (256B aligned): ~202 MB total.
  char* ws = (char*)d_ws;
  size_t off = 0;
  auto carve = [&](size_t bytes) {
    char* p = ws + off;
    off += (bytes + 255) & ~(size_t)255;
    return p;
  };
  bf16t* xbf = (bf16t*)carve((size_t)kM * kD * 2);
  bf16t* wih[4];
  bf16t* whh[4];
  for (int dl = 0; dl < 4; ++dl) wih[dl] = (bf16t*)carve((size_t)kG * kD * 2);
  for (int dl = 0; dl < 4; ++dl) whh[dl] = (bf16t*)carve((size_t)kG * kH * 2);
  bf16t* gin[2];
  for (int d = 0; d < 2; ++d) gin[d] = (bf16t*)carve((size_t)kM * kG * 2);
  bf16t* ybf[2];
  for (int d = 0; d < 2; ++d) ybf[d] = (bf16t*)carve((size_t)kM * kH * 2);
  bf16t* hping[2];
  for (int d = 0; d < 2; ++d) hping[d] = (bf16t*)carve((size_t)2 * kB * kH * 2);
  unsigned* bar = (unsigned*)carve(256);

  auto cvt = [&](const float* s, bf16t* dst, int n) {
    int grid = (n + 1023) / 1024;
    if (grid > 8192) grid = 8192;
    if (grid < 1) grid = 1;
    cvt_f32_bf16_kernel<<<grid, 256, 0, stream>>>(s, dst, n);
  };

  // One-time dtype conversions (params: f0,b0,f1,b1 at d_in[4+3*dl..]).
  cvt(x, xbf, kM * kD);
  for (int dl = 0; dl < 4; ++dl) {
    cvt((const float*)d_in[4 + 3 * dl], wih[dl], kG * kD);
    cvt((const float*)d_in[5 + 3 * dl], whh[dl], kG * kH);
  }

  const size_t HN_OFF = (size_t)kB * kT * (2 * kH);
  const size_t CN_OFF = HN_OFF + (size_t)2 * kB * (2 * kH);

  for (int layer = 0; layer < 2; ++layer) {
    // Seed h ping buffers with h0 and reset the inter-WG barrier counters.
    cvt(h0, hping[0], kB * kH);
    cvt(h0, hping[1], kB * kH);
    zero_u32_kernel<<<1, 64, 0, stream>>>(bar, 64);

    // Hoisted input projections for both directions (fully parallel GEMMs).
    for (int d = 0; d < 2; ++d) {
      const int dl = layer * 2 + d;  // f0=0, b0=1, f1=2, b1=3
      const bf16t* X = (layer == 0) ? xbf : ybf[d];
      gemm_xw_bias_kernel<<<4096, 256, 0, stream>>>(
          X, wih[dl], (const float*)d_in[6 + 3 * dl], gin[d]);
    }

    // Persistent bidirectional scan (both directions concurrent).
    RecArgs ra;
    ra.masks = masks; ra.h0 = h0; ra.c0 = c0;
    for (int d = 0; d < 2; ++d) {
      const int dl = layer * 2 + d;
      DirArgs& da = ra.dir[d];
      da.gin   = gin[d];
      da.whh   = whh[dl];
      da.hping = hping[d];
      da.ybf   = (layer == 0) ? ybf[d] : nullptr;
      da.yf32  = (layer == 1) ? (out + d * kH) : nullptr;
      da.hN = out + HN_OFF + (size_t)layer * kB * (2 * kH) + d * kH;
      da.cN = out + CN_OFF + (size_t)layer * kB * (2 * kH) + d * kH;
      da.bar = bar + d * 32;   // 128B apart
      da.reverse = d;          // d=0 forward, d=1 backward
    }
    lstm_scan_kernel<<<64, 128, 64 * 512 * 2, stream>>>(ra);
  }
}
